// VisionTemporalAttention_25383256720154
// MI455X (gfx1250) — matrix operand order repaired and checked by reading, not compiled, not run on hardware
//
#include <hip/hip_runtime.h>
#include <math.h>

// CDNA5 wave32 WMMA types
typedef __attribute__((ext_vector_type(16))) _Float16 v16h;
typedef __attribute__((ext_vector_type(2)))  _Float16 v2h;
typedef __attribute__((ext_vector_type(8)))  float    v8f;

#define S_TOK  4096   // total tokens B*T*N
#define DMODEL 512
#define NHEAD  8
#define HDIM   64

// seq_id per reference: searchsorted(cu[1:-1], pos, side='left')
// => number of interior boundaries strictly less than pos
__device__ __forceinline__ int seq_of(const int* __restrict__ cu, int ncu, int pos) {
    int sid = 0;
    for (int i = 1; i + 1 < ncu; ++i) sid += (pos > cu[i]) ? 1 : 0;
    return sid;
}

// ---------------------------------------------------------------------------
// Software-pipelined tiled GEMM on v_wmma_f32_16x16x32_f16.
//
// Block tile 128x64, K-step 32, 128 threads = 4 waves; each wave owns a
// 32x64 strip (2 A fragments x 4 B fragments = 8 accumulators).
//
// LDS tiles are double-buffered and kept in *fragment order*: each lane's
// 16 f16 elements are contiguous 32B, so a fragment load is one v16h read
// (2x ds_load_b128). Fragment-order inverse maps (from ISA VGPR layouts):
//   A (16x32): K(e,half) = (e<8 ? e : e+8) + 8*half
//      kk -> half=(kk>>3)&1, e=(kk&7)+8*(kk>>4), lane=(m&15)+16*half
//   B (32x16): K(e,half) = (e<8 ? e : e+8) + 8*half
//      kk -> half=(kk>>3)&1, e=(kk&7)+8*(kk>>4), lane=(n&15)+16*half
// Even kk keeps (kk,kk+1) adjacent in both maps => stage packed 2xf16.
//
// Pipeline: loads for tile t+1 are issued into registers BEFORE the WMMAs
// for tile t, converted/stored to the alternate LDS buffer after, with one
// __syncthreads per K-step.
//
// TRANS_B: B stored N x K (q @ k^T).  MASK: block-diag -inf (scores).
// ---------------------------------------------------------------------------
#define BM 128
#define BN 64
#define BK 32
#define JA (BM * BK / 2 / 128)   // 16 A pairs per thread
#define JB (BN * BK / 2 / 128)   // 8  B pairs per thread

template<bool TRANS_B, bool MASK>
__global__ __launch_bounds__(128)
void wmma_gemm(const float* __restrict__ A, const float* __restrict__ B,
               float* __restrict__ C,
               int K, int lda, int ldb, int ldc,
               long strideAh, long strideBh, long strideCh,
               float scale, const float* __restrict__ bias,
               const int* __restrict__ cu, int ncu)
{
    __shared__ __attribute__((aligned(32))) _Float16 Afrag[2][8][32][16]; // [buf][frag][lane][e]
    __shared__ __attribute__((aligned(32))) _Float16 Bfrag[2][4][32][16];

    const int h = blockIdx.z;
    A += (long)h * strideAh;
    B += (long)h * strideBh;
    C += (long)h * strideCh;

    const int row0  = blockIdx.y * BM;
    const int col0  = blockIdx.x * BN;
    const int tid   = threadIdx.x;
    const int wave  = tid >> 5;
    const int lane  = tid & 31;
    const int lrow  = lane & 15;
    const int lhalf = lane >> 4;

    float2 ra[JA];
    float2 rb[JB];

    // ---- issue global loads for tile at k0 into registers (batched, MLP)
    auto loadA = [&](int k0) {
        #pragma unroll
        for (int j = 0; j < JA; ++j) {
            const int p  = j * 128 + tid;
            const int m  = p >> 4;
            const int kk = (p & 15) * 2;
            ra[j] = *(const float2*)&A[(long)(row0 + m) * lda + (k0 + kk)];
        }
    };
    auto loadB = [&](int k0) {
        #pragma unroll
        for (int j = 0; j < JB; ++j) {
            const int p = j * 128 + tid;
            if (TRANS_B) {
                const int n  = p >> 4;
                const int kk = (p & 15) * 2;      // contiguous along k
                rb[j] = *(const float2*)&B[(long)(col0 + n) * ldb + (k0 + kk)];
            } else {
                const int kk = (p >> 6) * 2;
                const int n  = p & 63;            // coalesced along n
                rb[j].x = B[(long)(k0 + kk)     * ldb + (col0 + n)];
                rb[j].y = B[(long)(k0 + kk + 1) * ldb + (col0 + n)];
            }
        }
    };
    // ---- convert + store registers into fragment-order LDS buffer
    auto storeA = [&](int buf) {
        #pragma unroll
        for (int j = 0; j < JA; ++j) {
            const int p  = j * 128 + tid;
            const int m  = p >> 4;
            const int kk = (p & 15) * 2;
            v2h pk; pk[0] = (_Float16)ra[j].x; pk[1] = (_Float16)ra[j].y;
            const int half = (kk >> 3) & 1;
            const int e    = (kk & 7) + 8 * (kk >> 4);
            *(v2h*)&Afrag[buf][m >> 4][(m & 15) + 16 * half][e] = pk;
        }
    };
    auto storeB = [&](int buf) {
        #pragma unroll
        for (int j = 0; j < JB; ++j) {
            const int p = j * 128 + tid;
            int n, kk;
            if (TRANS_B) { n = p >> 4;  kk = (p & 15) * 2; }
            else         { n = p & 63;  kk = (p >> 6) * 2; }
            v2h pk; pk[0] = (_Float16)rb[j].x; pk[1] = (_Float16)rb[j].y;
            *(v2h*)&Bfrag[buf][n >> 4][(n & 15) + 16 * ((kk >> 3) & 1)][(kk & 7) + 8 * (kk >> 4)] = pk;
        }
    };

    v8f acc[2][4] = {};

    // ---- pipeline prologue
    loadA(0); loadB(0);
    storeA(0); storeB(0);
    __syncthreads();

    const int nt = K / BK;
    for (int t = 0; t < nt; ++t) {
        const int buf = t & 1;
        // issue next tile's global loads before consuming this tile
        if (t + 1 < nt) { loadA((t + 1) * BK); loadB((t + 1) * BK); }

        const v16h a0 = *(const v16h*)&Afrag[buf][2 * wave][lane][0];
        const v16h a1 = *(const v16h*)&Afrag[buf][2 * wave + 1][lane][0];
        #pragma unroll
        for (int j = 0; j < 4; ++j) {
            const v16h bf = *(const v16h*)&Bfrag[buf][j][lane][0];
            acc[0][j] = __builtin_amdgcn_wmma_f32_16x16x32_f16(
                            false, a0, false, bf, (short)0, acc[0][j], false, false);
            acc[1][j] = __builtin_amdgcn_wmma_f32_16x16x32_f16(
                            false, a1, false, bf, (short)0, acc[1][j], false, false);
        }

        if (t + 1 < nt) {
            storeA(buf ^ 1); storeB(buf ^ 1);
            __syncthreads();
        }
    }

    // ---- epilogue. C/D layout: VGPR r, lane -> M = r + 8*lhalf, N = lrow
    #pragma unroll
    for (int ar = 0; ar < 2; ++ar) {
        #pragma unroll
        for (int j = 0; j < 4; ++j) {
            #pragma unroll
            for (int r = 0; r < 8; ++r) {
                const int row = row0 + wave * 32 + ar * 16 + r + 8 * lhalf;
                const int col = col0 + j * 16 + lrow;
                float v = acc[ar][j][r] * scale;
                if (bias) v += bias[col];
                if (MASK) {
                    if (seq_of(cu, ncu, row) != seq_of(cu, ncu, col))
                        v = -__builtin_inff();
                }
                C[(long)row * ldc + col] = v;
            }
        }
    }
}

// ---------------------------------------------------------------------------
// RoPE in place on q and k slices of packed qkv [S][3*DMODEL].
// ---------------------------------------------------------------------------
__global__ __launch_bounds__(256)
void rope_kernel(float* __restrict__ qkv)
{
    const int idx = blockIdx.x * blockDim.x + threadIdx.x;
    if (idx >= S_TOK * NHEAD * (HDIM / 2)) return;
    const int i = idx & 31;          // freq index 0..31
    const int h = (idx >> 5) & 7;    // head
    const int s = idx >> 8;          // token

    const float inv_freq = __expf(-((2.0f * i) / (float)HDIM) * __logf(10000.0f));
    const float ang = (float)s * inv_freq;
    const float c = cosf(ang), sn = sinf(ang);

    float* qb = qkv + (long)s * (3 * DMODEL) + h * HDIM;          // q
    float* kb = qb + DMODEL;                                      // k
    const float qa = qb[i], qbv = qb[i + 32];
    qb[i]      = qa * c - qbv * sn;
    qb[i + 32] = qbv * c + qa * sn;
    const float ka = kb[i], kbv = kb[i + 32];
    kb[i]      = ka * c - kbv * sn;
    kb[i + 32] = kbv * c + ka * sn;
}

// ---------------------------------------------------------------------------
// Row softmax in place over attn rows of length S_TOK (fp32, matches ref).
// ---------------------------------------------------------------------------
__global__ __launch_bounds__(256)
void softmax_kernel(float* __restrict__ attn)
{
    __shared__ float red[256];
    float* p = attn + (long)blockIdx.x * S_TOK;
    const int t = threadIdx.x;

    float m = -__builtin_inff();
    for (int i = t; i < S_TOK; i += 256) m = fmaxf(m, p[i]);
    red[t] = m; __syncthreads();
    for (int s = 128; s > 0; s >>= 1) {
        if (t < s) red[t] = fmaxf(red[t], red[t + s]);
        __syncthreads();
    }
    m = red[0]; __syncthreads();

    float sum = 0.0f;
    for (int i = t; i < S_TOK; i += 256) sum += __expf(p[i] - m);
    red[t] = sum; __syncthreads();
    for (int s = 128; s > 0; s >>= 1) {
        if (t < s) red[t] += red[t + s];
        __syncthreads();
    }
    const float inv = 1.0f / red[0];

    for (int i = t; i < S_TOK; i += 256) p[i] = __expf(p[i] - m) * inv;
}

// ---------------------------------------------------------------------------
// Launch: qkv GEMM -> RoPE -> masked scores -> softmax -> attn@v -> proj
// d_out = [ final (4096*512 f32) | attn (8*4096*4096 f32) ]
// d_ws  = [ qkv (4096*1536 f32) | ctx (4096*512 f32) ]  = 32 MB
// ---------------------------------------------------------------------------
extern "C" void kernel_launch(void* const* d_in, const int* in_sizes, int n_in,
                              void* d_out, int out_size, void* d_ws, size_t ws_size,
                              hipStream_t stream)
{
    const float* x      = (const float*)d_in[0];   // (1,16,256,512) -> 4096x512
    const float* qkv_w  = (const float*)d_in[1];   // 512x1536
    const float* qkv_b  = (const float*)d_in[2];   // 1536
    const float* proj_w = (const float*)d_in[3];   // 512x512
    const float* proj_b = (const float*)d_in[4];   // 512
    const int*   cu     = (const int*)d_in[5];
    const int    ncu    = in_sizes[5];

    float* out_final = (float*)d_out;
    float* attn      = out_final + (long)S_TOK * DMODEL;

    float* qkv = (float*)d_ws;
    float* ctx = qkv + (long)S_TOK * 3 * DMODEL;

    const dim3 blk(128);

    // 1) qkv = x @ qkv_w + qkv_b           (4096x1536, K=512)
    wmma_gemm<false, false><<<dim3(1536 / BN, S_TOK / BM, 1), blk, 0, stream>>>(
        x, qkv_w, qkv, 512, 512, 1536, 1536,
        0, 0, 0, 1.0f, qkv_b, nullptr, 0);

    // 2) RoPE on q,k in place
    rope_kernel<<<(S_TOK * NHEAD * 32 + 255) / 256, 256, 0, stream>>>(qkv);

    // 3) scores = scale * q @ k^T, block-diag mask   (per head: 4096x4096, K=64)
    wmma_gemm<true, true><<<dim3(S_TOK / BN, S_TOK / BM, NHEAD), blk, 0, stream>>>(
        qkv /*q*/, qkv + DMODEL /*k*/, attn, HDIM, 3 * DMODEL, 3 * DMODEL, S_TOK,
        HDIM, HDIM, (long)S_TOK * S_TOK, 0.125f /*hd^-0.5*/, nullptr, cu, ncu);

    // 4) row softmax in place (fp32, as reference)
    softmax_kernel<<<NHEAD * S_TOK, 256, 0, stream>>>(attn);

    // 5) ctx = attn @ v                     (per head: 4096x64, K=4096)
    wmma_gemm<false, false><<<dim3(1, S_TOK / BM, NHEAD), blk, 0, stream>>>(
        attn, qkv + 2 * DMODEL /*v*/, ctx, S_TOK, S_TOK, 3 * DMODEL, DMODEL,
        (long)S_TOK * S_TOK, HDIM, HDIM, 1.0f, nullptr, nullptr, 0);

    // 6) final = ctx @ proj_w + proj_b      (4096x512, K=512)
    wmma_gemm<false, false><<<dim3(DMODEL / BN, S_TOK / BM, 1), blk, 0, stream>>>(
        ctx, proj_w, out_final, 512, 512, 512, 512,
        0, 0, 0, 1.0f, proj_b, nullptr, 0);
}
